// BertBiAttention_15676630630433
// MI455X (gfx1250) — compile-verified, matrix operand order checked by reading, not run on hardware
//
#include <hip/hip_runtime.h>
#include <hip/hip_bf16.h>

// MI455X / gfx1250: wave32, v_wmma_f32_16x16x32_bf16 (f32 accumulate).

typedef __attribute__((ext_vector_type(2)))  __bf16 v2bf;
typedef __attribute__((ext_vector_type(4)))  __bf16 v4bf;
typedef __attribute__((ext_vector_type(8)))  __bf16 v8bf;
typedef __attribute__((ext_vector_type(16))) __bf16 v16bf;
typedef __attribute__((ext_vector_type(4)))  float  v4f;
typedef __attribute__((ext_vector_type(8)))  float  v8f;

#define Lseq 512
#define Dm   1024
#define Hh   16
#define DHq  64
#define DKVH 32
#define Bb   8
#define ATT_SCALE 0.125f   // 1/sqrt(64)

#if defined(__has_builtin)
#if __has_builtin(__builtin_amdgcn_cvt_pk_bf16_f32)
#define HAS_PK_BF16 1
#endif
#if __has_builtin(__builtin_amdgcn_perm)
#define HAS_PERM 1
#endif
#endif

// pack two f32 -> two bf16 (RNE). With v_perm_b32: 2x v_add3 + 1x v_perm.
__device__ __forceinline__ v2bf pkbf(float a, float b) {
#ifdef HAS_PK_BF16
  auto t = __builtin_amdgcn_cvt_pk_bf16_f32(a, b);
  return __builtin_bit_cast(v2bf, t);
#else
  unsigned ua = __builtin_bit_cast(unsigned, a);
  unsigned ub = __builtin_bit_cast(unsigned, b);
  ua += 0x7fffu + ((ua >> 16) & 1u);
  ub += 0x7fffu + ((ub >> 16) & 1u);
#ifdef HAS_PERM
  return __builtin_bit_cast(v2bf,
      __builtin_amdgcn_perm(ub, ua, 0x07060302u));   // {ub.hi16, ua.hi16}
#else
  unsigned packed = (ua >> 16) | (ub & 0xffff0000u);
  return __builtin_bit_cast(v2bf, packed);
#endif
#endif
}

__device__ __forceinline__ v4bf  cat2(v2bf a, v2bf b) {
  return __builtin_shufflevector(a, b, 0, 1, 2, 3);
}
__device__ __forceinline__ v8bf  cat4(v4bf a, v4bf b) {
  return __builtin_shufflevector(a, b, 0, 1, 2, 3, 4, 5, 6, 7);
}
__device__ __forceinline__ v16bf cat8(v8bf a, v8bf b) {
  return __builtin_shufflevector(a, b, 0, 1, 2, 3, 4, 5, 6, 7,
                                       8, 9, 10, 11, 12, 13, 14, 15);
}
// 8 contiguous bf16 -> one b128 load (global or LDS)
__device__ __forceinline__ v8bf ld8(const __bf16* p) {
  return *reinterpret_cast<const v8bf*>(p);
}

// ---------------------------------------------------------------------------
// One-time f32 -> bf16 cast of activations X (coalesced, 4 elems/thread).
// Removes all conversion VALU from the projection inner loop and halves
// X read bandwidth (each X row is re-read by every n-block).
// ---------------------------------------------------------------------------
__global__ __launch_bounds__(256)
void xcast(const float* __restrict__ X, __bf16* __restrict__ Xb) {
  const int idx = blockIdx.x * 256 + threadIdx.x;       // 4-element groups
  const v4f x = *reinterpret_cast<const v4f*>(X + (size_t)idx * 4);
  *reinterpret_cast<v4bf*>(Xb + (size_t)idx * 4) =
      cat2(pkbf(x[0], x[1]), pkbf(x[2], x[3]));
}

// ---------------------------------------------------------------------------
// Weight transpose + bf16 cast:  W[K=1024][N] (f32) -> Wt[N][1024] (bf16).
// ---------------------------------------------------------------------------
template <int NSHIFT>
__global__ __launch_bounds__(256)
void wtrans(const float* __restrict__ W, __bf16* __restrict__ Wt) {
  constexpr int N = 1 << NSHIFT;
  const int idx = blockIdx.x * 256 + threadIdx.x;       // over N*512 k-pairs
  const int n  = idx & (N - 1);                         // coalesced read dim
  const int k2 = (idx >> NSHIFT) * 2;
  *reinterpret_cast<v2bf*>(Wt + (size_t)n * Dm + k2) =
      pkbf(W[k2 * N + n], W[(k2 + 1) * N + n]);
}

// ---------------------------------------------------------------------------
// Stage 1: Y[4096, N] = bf16( Xb[4096,1024] @ Wt^T + bias ), all bf16 inputs.
// One wave -> 16x64 output tile (4 accumulators), 32 k-steps.
// Every fragment = two contiguous 8-element runs -> 2x global_load_b128.
// MODE 0: Q -> [b,h,l,64]; MODE 1: K -> [b,h,l,32]; MODE 2: V^T -> [b,h,32,l]
// ---------------------------------------------------------------------------
template <int MODE>
__global__ __launch_bounds__(32)
void proj_wmma(const __bf16* __restrict__ Xb, const __bf16* __restrict__ Wt,
               const float* __restrict__ bias, __bf16* __restrict__ Y) {
  const int lane = threadIdx.x;
  const int lr = lane & 15;
  const int lh = lane >> 4;
  const int n0 = blockIdx.x * 64;
  const int m0 = blockIdx.y * 16;
  const int m  = m0 + lr;

  v8f acc[4] = {{}, {}, {}, {}};
  for (int kk = 0; kk < Dm / 32; ++kk) {
    const __bf16* xp = Xb + (size_t)m * Dm + kk * 32;
    const v16bf a = cat8(ld8(xp + lh * 8), ld8(xp + 16 + lh * 8));
#pragma unroll
    for (int nt = 0; nt < 4; ++nt) {
      const int n = n0 + nt * 16 + lr;
      const __bf16* wp = Wt + (size_t)n * Dm + kk * 32;
      const v16bf bfrag = cat8(ld8(wp + lh * 8), ld8(wp + 16 + lh * 8));
      acc[nt] = __builtin_amdgcn_wmma_f32_16x16x32_bf16(
          false, a, false, bfrag, (short)0, acc[nt], false, false);
    }
  }

  const int bidx  = m0 >> 9;      // 16-row tile never crosses a token-512 boundary
  const int lpos0 = (m0 & 511) + lh * 8;
#pragma unroll
  for (int nt = 0; nt < 4; ++nt) {
    const int n = n0 + nt * 16 + lr;
    const float bvn = bias[n];
    if (MODE == 2) {              // V^T: lane's column is row-contiguous -> b128
      const int h = n >> 5, f = n & 31;
      const size_t idx = ((size_t)(bidx * Hh + h) * DKVH + f) * Lseq + lpos0;
      v8bf o = cat4(cat2(pkbf(acc[nt][0] + bvn, acc[nt][1] + bvn),
                         pkbf(acc[nt][2] + bvn, acc[nt][3] + bvn)),
                    cat2(pkbf(acc[nt][4] + bvn, acc[nt][5] + bvn),
                         pkbf(acc[nt][6] + bvn, acc[nt][7] + bvn)));
      *reinterpret_cast<v8bf*>(Y + idx) = o;
    } else {
#pragma unroll
      for (int v = 0; v < 8; ++v) {
        const int lpos = lpos0 + v;
        size_t idx;
        if (MODE == 0) {          // Q: [b,h,l,64]
          const int h = n >> 6, f = n & 63;
          idx = ((size_t)(bidx * Hh + h) * Lseq + lpos) * DHq + f;
        } else {                  // K: [b,h,l,32]
          const int h = n >> 5, f = n & 31;
          idx = ((size_t)(bidx * Hh + h) * Lseq + lpos) * DKVH + f;
        }
        v2bf o = pkbf(acc[nt][v] + bvn, 0.0f);
        Y[idx] = o[0];
      }
    }
  }
}

// ---------------------------------------------------------------------------
// Stage 2: one wave per (b, h, 16-query tile).
// scores -> LDS(f32), rowmax via shfl, softmax vectorized float4,
// probs -> d_out (b128 stores) + bf16 copy in LDS, ctx = P @ [Va|Vb] via WMMA.
// ---------------------------------------------------------------------------
template <bool USE_MASK>
__global__ __launch_bounds__(32)
void attn_wmma(const __bf16* __restrict__ Q,
               const __bf16* __restrict__ Ka, const __bf16* __restrict__ Kb,
               const __bf16* __restrict__ Va, const __bf16* __restrict__ Vb,
               const float* __restrict__ ma, const float* __restrict__ mb,
               float* __restrict__ ctx, float* __restrict__ probs) {
  __shared__ __align__(16) float  sS[16 * Lseq];   // 32 KB
  __shared__ __align__(16) __bf16 sP[16 * Lseq];   // 16 KB
  __shared__ float sRed[16];

  const int lane = threadIdx.x;
  const int lr = lane & 15, lh = lane >> 4;
  const int qt = blockIdx.x;
  const int bh = blockIdx.y;
  const int b  = bh >> 4, h = bh & 15;
  const int q0 = qt * 16;

  const __bf16* Qp  = Q  + (size_t)bh * Lseq * DHq;
  const __bf16* Kap = Ka + (size_t)bh * Lseq * DKVH;
  const __bf16* Kbp = Kb + (size_t)bh * Lseq * DKVH;
  const __bf16* Vap = Va + (size_t)bh * DKVH * Lseq;
  const __bf16* Vbp = Vb + (size_t)bh * DKVH * Lseq;

  // Q fragments (features 0-31 | 32-63): 2 b128 loads each
  const __bf16* Qrow = Qp + (q0 + lr) * DHq;
  const v16bf aq0 = cat8(ld8(Qrow + lh * 8),      ld8(Qrow + 16 + lh * 8));
  const v16bf aq1 = cat8(ld8(Qrow + 32 + lh * 8), ld8(Qrow + 48 + lh * 8));

  float pm[8];
#pragma unroll
  for (int v = 0; v < 8; ++v) pm[v] = -3.0e38f;

  // ---- scores ----
  for (int kt = 0; kt < Lseq / 16; ++kt) {
    const int t = kt * 16 + lr;               // key token (N) for this lane
    const __bf16* ka = Kap + t * DKVH;
    const __bf16* kb = Kbp + t * DKVH;
    const v16bf bka = cat8(ld8(ka + lh * 8), ld8(ka + 16 + lh * 8));
    const v16bf bkb = cat8(ld8(kb + lh * 8), ld8(kb + 16 + lh * 8));
    v8f acc = {};
    acc = __builtin_amdgcn_wmma_f32_16x16x32_bf16(false, aq0, false, bka,
                                                  (short)0, acc, false, false);
    acc = __builtin_amdgcn_wmma_f32_16x16x32_bf16(false, aq1, false, bkb,
                                                  (short)0, acc, false, false);
    float bmask = 0.0f;
    if (USE_MASK) bmask = ma[b * Lseq + t] + mb[b * Lseq + t];
#pragma unroll
    for (int v = 0; v < 8; ++v) {
      const float s = acc[v] * ATT_SCALE + bmask;
      sS[(lh * 8 + v) * Lseq + t] = s;        // C layout: half owns 8 full rows
      pm[v] = fmaxf(pm[v], s);
    }
  }
#pragma unroll
  for (int v = 0; v < 8; ++v) {
#pragma unroll
    for (int off = 1; off < 16; off <<= 1)
      pm[v] = fmaxf(pm[v], __shfl_xor(pm[v], off, 32));
  }
  if (lr == 0) {
#pragma unroll
    for (int v = 0; v < 8; ++v) sRed[lh * 8 + v] = pm[v];
  }
  __syncthreads();

  // ---- softmax: lane -> row (lane&15), key half (lane>>4); float4 vectorized
  {
    const int row = lr;
    const float mx = sRed[row];
    v4f* rS = reinterpret_cast<v4f*>(sS + row * Lseq + lh * 256);
    float sum = 0.0f;
    for (int i = 0; i < 64; ++i) {
      v4f x = rS[i];
      x[0] = __expf(x[0] - mx); x[1] = __expf(x[1] - mx);
      x[2] = __expf(x[2] - mx); x[3] = __expf(x[3] - mx);
      sum += (x[0] + x[1]) + (x[2] + x[3]);
      rS[i] = x;
    }
    const float total = sum + __shfl_xor(sum, 16, 32);
    const float inv = 1.0f / total;
    float* pout = probs + ((size_t)bh * Lseq + (q0 + row)) * Lseq + lh * 256;
    v4f*  pg = reinterpret_cast<v4f*>(pout);
    v4bf* rP = reinterpret_cast<v4bf*>(sP + row * Lseq + lh * 256);
    for (int i = 0; i < 64; ++i) {
      const v4f e = rS[i];
      v4f p; p[0] = e[0] * inv; p[1] = e[1] * inv;
             p[2] = e[2] * inv; p[3] = e[3] * inv;
      pg[i] = p;                               // f32 probs: global_store_b128
      rP[i] = cat2(pkbf(p[0], p[1]), pkbf(p[2], p[3]));
    }
  }
  __syncthreads();

  // ---- ctx = P @ [Va|Vb]: N=64 -> 4 tiles, K=512 -> 16 steps ----
  v8f cacc[4] = {{}, {}, {}, {}};
  for (int kt = 0; kt < Lseq / 32; ++kt) {
    const __bf16* prow = sP + lr * Lseq + kt * 32;
    const v16bf ap = cat8(ld8(prow + lh * 8), ld8(prow + 16 + lh * 8));  // ds b128
#pragma unroll
    for (int nt = 0; nt < 4; ++nt) {
      const int F = nt * 16 + lr;
      const __bf16* Vp = (F < 32) ? (Vap + (size_t)F * Lseq)
                                  : (Vbp + (size_t)(F - 32) * Lseq);
      const __bf16* vrow = Vp + kt * 32;
      const v16bf bvf = cat8(ld8(vrow + lh * 8), ld8(vrow + 16 + lh * 8));
      cacc[nt] = __builtin_amdgcn_wmma_f32_16x16x32_bf16(
          false, ap, false, bvf, (short)0, cacc[nt], false, false);
    }
  }
#pragma unroll
  for (int nt = 0; nt < 4; ++nt) {
#pragma unroll
    for (int v = 0; v < 8; ++v) {
      const int qr = q0 + lh * 8 + v;
      const int F  = nt * 16 + lr;
      ctx[((size_t)b * Lseq + qr) * Dm + h * DHq + F] = cacc[nt][v];
    }
  }
}

// ---------------------------------------------------------------------------
extern "C" void kernel_launch(void* const* d_in, const int* in_sizes, int n_in,
                              void* d_out, int out_size, void* d_ws, size_t ws_size,
                              hipStream_t stream) {
  const float* x[3]   = {(const float*)d_in[0],  (const float*)d_in[2],  (const float*)d_in[4]};
  const float* msk[3] = {(const float*)d_in[1],  (const float*)d_in[3],  (const float*)d_in[5]};
  const float* wq[3]  = {(const float*)d_in[6],  (const float*)d_in[12], (const float*)d_in[18]};
  const float* bq[3]  = {(const float*)d_in[7],  (const float*)d_in[13], (const float*)d_in[19]};
  const float* wk[3]  = {(const float*)d_in[8],  (const float*)d_in[14], (const float*)d_in[20]};
  const float* bk[3]  = {(const float*)d_in[9],  (const float*)d_in[15], (const float*)d_in[21]};
  const float* wv[3]  = {(const float*)d_in[10], (const float*)d_in[16], (const float*)d_in[22]};
  const float* bv[3]  = {(const float*)d_in[11], (const float*)d_in[17], (const float*)d_in[23]};

  // ws (bf16): Q x3, K x3, V^T x3, Xb x3, Wt_q x3, Wt_k/Wt_v x3  (~87 MB)
  __bf16* ws = (__bf16*)d_ws;
  const size_t QSZ  = (size_t)Bb * Hh * Lseq * DHq;    // 4,194,304
  const size_t KSZ  = (size_t)Bb * Hh * Lseq * DKVH;   // 2,097,152
  const size_t XSZ  = (size_t)Bb * Lseq * Dm;          // 4,194,304
  const size_t WQSZ = (size_t)Dm * Dm;                 // 1,048,576
  const size_t WKSZ = (size_t)512 * Dm;                //   524,288
  __bf16 *Qm[3], *Km[3], *Vm[3], *Xb[3], *Wtq[3], *Wtk[3], *Wtv[3];
  size_t off = 0;
  for (int i = 0; i < 3; ++i) { Qm[i]  = ws + off; off += QSZ;  }
  for (int i = 0; i < 3; ++i) { Km[i]  = ws + off; off += KSZ;  }
  for (int i = 0; i < 3; ++i) { Vm[i]  = ws + off; off += KSZ;  }
  for (int i = 0; i < 3; ++i) { Xb[i]  = ws + off; off += XSZ;  }
  for (int i = 0; i < 3; ++i) { Wtq[i] = ws + off; off += WQSZ; }
  for (int i = 0; i < 3; ++i) { Wtk[i] = ws + off; off += WKSZ; }
  for (int i = 0; i < 3; ++i) { Wtv[i] = ws + off; off += WKSZ; }

  const dim3 blk32(32, 1, 1);
  const dim3 blk256(256, 1, 1);
  for (int i = 0; i < 3; ++i) {
    xcast<<<dim3(4096), blk256, 0, stream>>>(x[i], Xb[i]);
    wtrans<10><<<dim3(Dm * 2), blk256, 0, stream>>>(wq[i], Wtq[i]);
    wtrans<9><<<dim3(512 * 2), blk256, 0, stream>>>(wk[i], Wtk[i]);
    wtrans<9><<<dim3(512 * 2), blk256, 0, stream>>>(wv[i], Wtv[i]);
  }
  for (int i = 0; i < 3; ++i) {
    proj_wmma<0><<<dim3(Dm / 64, 256), blk32, 0, stream>>>(Xb[i], Wtq[i], bq[i], Qm[i]);
    proj_wmma<1><<<dim3(512 / 64, 256), blk32, 0, stream>>>(Xb[i], Wtk[i], bk[i], Km[i]);
    proj_wmma<2><<<dim3(512 / 64, 256), blk32, 0, stream>>>(Xb[i], Wtv[i], bv[i], Vm[i]);
  }

  float* out = (float*)d_out;
  const size_t CSZ = (size_t)Bb * Lseq * Dm;          // 4,194,304
  const size_t PSZ = (size_t)Bb * Hh * Lseq * Lseq;   // 33,554,432
  float* c2 = out;                  // return order: c2, c1, c3, p2, p1, p3
  float* c1 = out + CSZ;
  float* c3 = out + 2 * CSZ;
  float* p2 = out + 3 * CSZ;
  float* p1 = p2 + PSZ;
  float* p3 = p2 + 2 * PSZ;

  const dim3 agrid(Lseq / 16, Bb * Hh);
  // (c2,p2): Q1, KV 2+3, no mask bias
  attn_wmma<false><<<agrid, blk32, 0, stream>>>(Qm[0], Km[1], Km[2], Vm[1], Vm[2],
                                                msk[0], msk[0], c2, p2);
  // (c1,p1): Q2, KV 1+3, bias m1+m3
  attn_wmma<true><<<agrid, blk32, 0, stream>>>(Qm[1], Km[0], Km[2], Vm[0], Vm[2],
                                               msk[0], msk[2], c1, p1);
  // (c3,p3): Q3, KV 1+2, bias m1+m2
  attn_wmma<true><<<agrid, blk32, 0, stream>>>(Qm[2], Km[0], Km[1], Vm[0], Vm[1],
                                               msk[0], msk[1], c3, p3);
}
